// DGDNN_21199958573634
// MI455X (gfx1250) — matrix operand order repaired and checked by reading, not compiled, not run on hardware
//
#include <hip/hip_runtime.h>
#include <hip/hip_bf16.h>
#include <math.h>

#define NN 8192
#define FF 64
#define KSPLIT 8
#define KCHUNK (NN / KSPLIT)
#define WSLICES 32

typedef __attribute__((ext_vector_type(16))) __bf16 v16bf;
typedef __attribute__((ext_vector_type(8)))  float  v8f;
typedef __attribute__((ext_vector_type(4)))  float  f32x4;
typedef __attribute__((ext_vector_type(16))) float  f32x16;

__device__ __forceinline__ __bf16 f32_to_bf16(float f) {
  unsigned u = __builtin_bit_cast(unsigned, f);
  u += 0x7FFFu + ((u >> 16) & 1u);            // round-to-nearest-even
  unsigned short h = (unsigned short)(u >> 16);
  return __builtin_bit_cast(__bf16, h);
}

// ---------- 1) w partials: wpart[slice][j] = sum_{i in slice} theta[i]*T[i][j]
__global__ __launch_bounds__(256) void wsum_part_kernel(const float* __restrict__ T,
                                                        const float* __restrict__ theta,
                                                        float* __restrict__ wpart) {
  int j  = blockIdx.x * 256 + threadIdx.x;
  int sl = blockIdx.y;
  int i0 = sl * (NN / WSLICES);
  const float* p = T + (size_t)i0 * NN + j;
  float s = 0.f;
  for (int i = 0; i < NN / WSLICES; ++i) {
    float t = __builtin_nontemporal_load(&p[(size_t)i * NN]);  // T streamed once: NT
    s = fmaf(theta[i0 + i], t, s);
  }
  wpart[(size_t)sl * NN + j] = s;
}

__global__ __launch_bounds__(256) void wreduce_kernel(const float* __restrict__ wpart,
                                                      float* __restrict__ w) {
  int j = blockIdx.x * 256 + threadIdx.x;
  float s = 0.f;
#pragma unroll
  for (int sl = 0; sl < WSLICES; ++sl) s += wpart[(size_t)sl * NN + j];
  w[j] = s;
}

// ---------- 2) XsT[f][i] = bf16(w[i] * X[i][f])   (transposed, bf16, 1 MB)
__global__ __launch_bounds__(256) void scale_transpose_kernel(const float* __restrict__ X,
                                                              const float* __restrict__ w,
                                                              __bf16* __restrict__ XsT) {
  int idx = blockIdx.x * 256 + threadIdx.x;   // idx = f*NN + i
  int f = idx >> 13;
  int i = idx & (NN - 1);
  XsT[idx] = f32_to_bf16(w[i] * X[(size_t)i * FF + f]);
}

// ---------- 3) Zpart[ks] = A[:, kslice] @ Xs[kslice, :]  via v_wmma_f32_16x16x32_bf16
__global__ __launch_bounds__(256) void gemm_wmma_kernel(const float* __restrict__ A,
                                                        const __bf16* __restrict__ XsT,
                                                        float* __restrict__ Zpart) {
  const int lane = threadIdx.x & 31;
  const int wave = threadIdx.x >> 5;
  const int half = lane >> 4;       // 0: lanes 0-15, 1: lanes 16-31
  const int l16  = lane & 15;
  const int m0 = (blockIdx.x * 8 + wave) * 16;       // 16-row tile
  const int k0 = blockIdx.y * KCHUNK;
  const float* Arow = A + (size_t)(m0 + l16) * NN;   // lane's A row (M = l16)

  const __bf16* B0 = XsT + (size_t)(l16 +  0) * NN;  // lane's B column per n-block
  const __bf16* B1 = XsT + (size_t)(l16 + 16) * NN;
  const __bf16* B2 = XsT + (size_t)(l16 + 32) * NN;
  const __bf16* B3 = XsT + (size_t)(l16 + 48) * NN;

  v8f acc0 = {}, acc1 = {}, acc2 = {}, acc3 = {};

  for (int k = k0; k < k0 + KCHUNK; k += 32) {
    // A fragment: 16 contiguous floats (K = k+16*half .. +15) -> bf16, matching B.
    // A is streamed exactly once -> non-temporal loads (don't rinse L2 of XsT/Zpart).
    const f32x4* p0 = reinterpret_cast<const f32x4*>(Arow + k + half * 16);
    const f32x4* p1 = reinterpret_cast<const f32x4*>(Arow + k + half * 16 + 8);
    f32x4 t0 = __builtin_nontemporal_load(p0);
    f32x4 t1 = __builtin_nontemporal_load(p0 + 1);
    f32x4 t2 = __builtin_nontemporal_load(p1);
    f32x4 t3 = __builtin_nontemporal_load(p1 + 1);
    f32x16 av;
#pragma unroll
    for (int e = 0; e < 4; ++e) {
      av[e]      = t0[e];
      av[4 + e]  = t1[e];
      av[8 + e]  = t2[e];
      av[12 + e] = t3[e];
    }
    // Vector fptrunc: lets ISel pick native packed f32->bf16 cvt if gfx1250 has it.
    v16bf a = __builtin_convertvector(av, v16bf);

    // B fragments: 16 contiguous bf16 per lane (column l16, K = kb..kb+15);
    // XsT is 1 MB and reused by every row tile -> keep regular-temporal (L2 resident).
    const int kb = k + half * 16;
    v16bf b0 = *reinterpret_cast<const v16bf*>(B0 + kb);
    v16bf b1 = *reinterpret_cast<const v16bf*>(B1 + kb);
    v16bf b2 = *reinterpret_cast<const v16bf*>(B2 + kb);
    v16bf b3 = *reinterpret_cast<const v16bf*>(B3 + kb);

    acc0 = __builtin_amdgcn_wmma_f32_16x16x32_bf16(false, a, false, b0, (short)0, acc0, false, false);
    acc1 = __builtin_amdgcn_wmma_f32_16x16x32_bf16(false, a, false, b1, (short)0, acc1, false, false);
    acc2 = __builtin_amdgcn_wmma_f32_16x16x32_bf16(false, a, false, b2, (short)0, acc2, false, false);
    acc3 = __builtin_amdgcn_wmma_f32_16x16x32_bf16(false, a, false, b3, (short)0, acc3, false, false);
  }

  // D layout: VGPR r, lane L -> M = r + 8*half, N = l16 (+16c)
  float* Zp = Zpart + (size_t)blockIdx.y * ((size_t)NN * FF);
  const int rbase = m0 + half * 8;
#pragma unroll
  for (int r = 0; r < 8; ++r) {
    float* zp = Zp + (size_t)(rbase + r) * FF + l16;
    zp[0]  = acc0[r];
    zp[16] = acc1[r];
    zp[32] = acc2[r];
    zp[48] = acc3[r];
  }
}

// ---------- 4) tail MLP + softmax, weights staged in LDS (stride-65 padding)
__global__ __launch_bounds__(256) void tail_kernel(
    const float* __restrict__ Zpart,
    const float* __restrict__ Wd,  const float* __restrict__ bd,
    const float* __restrict__ Wnf, const float* __restrict__ bnf,
    const float* __restrict__ Wm,  const float* __restrict__ bm,
    const float* __restrict__ Wr0, const float* __restrict__ br0,
    const float* __restrict__ Wr1, const float* __restrict__ br1,
    float* __restrict__ out) {
  __shared__ float sWd[64 * 65], sWnf[64 * 65], sWm[64 * 65], sWr0[64 * 65];
  __shared__ float sbd[64], sbnf[64], sbm[64], sbr0[64];
  __shared__ float sWr1[2 * 64];
  __shared__ float sbr1[2];
  __shared__ float bufA[64 * 64], bufB[64 * 64];

  const int tid = threadIdx.x;
  for (int idx = tid; idx < 64 * 64; idx += 256) {
    int o = idx >> 6, kk = idx & 63;
    sWd [o * 65 + kk] = Wd [idx];
    sWnf[o * 65 + kk] = Wnf[idx];
    sWm [o * 65 + kk] = Wm [idx];
    sWr0[o * 65 + kk] = Wr0[idx];
  }
  if (tid < 64) { sbd[tid] = bd[tid]; sbnf[tid] = bnf[tid]; sbm[tid] = bm[tid]; sbr0[tid] = br0[tid]; }
  if (tid < 128) sWr1[tid] = Wr1[tid];
  if (tid < 2)   sbr1[tid] = br1[tid];

  const int row0 = blockIdx.x * 64;
  for (int idx = tid; idx < 64 * 64; idx += 256) {
    size_t base = (size_t)row0 * FF + idx;
    float s = 0.f;
#pragma unroll
    for (int sl = 0; sl < KSPLIT; ++sl) s += Zpart[(size_t)sl * ((size_t)NN * FF) + base];
    bufA[idx] = s;
  }
  __syncthreads();

  auto layer = [&](const float* in, float* ob, const float* W, const float* b, int act) {
    for (int e = tid; e < 64 * 64; e += 256) {
      int r = e >> 6, o = e & 63;
      const float* ir = in + r * 64;
      const float* wr = W + o * 65;
      float s = b[o];
#pragma unroll
      for (int kk = 0; kk < 64; ++kk) s = fmaf(ir[kk], wr[kk], s);
      if (act == 1)      s = (s > 0.f) ? s : 0.01f * s;  // leaky_relu
      else if (act == 2) s = (s > 0.f) ? s : 0.f;        // relu
      ob[e] = s;
    }
    __syncthreads();
  };

  layer(bufA, bufB, sWd,  sbd,  1);   // diffusion + leaky_relu
  layer(bufB, bufA, sWnf, sbnf, 0);   // node_feature (no act)
  layer(bufA, bufB, sWm,  sbm,  2);   // model + relu
  layer(bufB, bufA, sWr0, sbr0, 2);   // readout[0] + relu

  for (int r = tid; r < 64; r += 256) {
    const float* fr = bufA + r * 64;
    float l0 = sbr1[0], l1 = sbr1[1];
#pragma unroll
    for (int kk = 0; kk < 64; ++kk) {
      l0 = fmaf(fr[kk], sWr1[kk], l0);
      l1 = fmaf(fr[kk], sWr1[64 + kk], l1);
    }
    float m = fmaxf(l0, l1);
    float e0 = __expf(l0 - m), e1 = __expf(l1 - m);
    float inv = 1.f / (e0 + e1);
    out[(size_t)(row0 + r) * 2 + 0] = e0 * inv;
    out[(size_t)(row0 + r) * 2 + 1] = e1 * inv;
  }
}

extern "C" void kernel_launch(void* const* d_in, const int* in_sizes, int n_in,
                              void* d_out, int out_size, void* d_ws, size_t ws_size,
                              hipStream_t stream) {
  (void)in_sizes; (void)n_in; (void)out_size; (void)ws_size;
  const float* X     = (const float*)d_in[0];
  const float* A     = (const float*)d_in[1];
  const float* T     = (const float*)d_in[2];
  const float* theta = (const float*)d_in[3];
  const float* Wd    = (const float*)d_in[4];
  const float* bd    = (const float*)d_in[5];
  const float* Wnf   = (const float*)d_in[6];
  const float* bnf   = (const float*)d_in[7];
  const float* Wm    = (const float*)d_in[8];
  const float* bm    = (const float*)d_in[9];
  const float* Wr0   = (const float*)d_in[10];
  const float* br0   = (const float*)d_in[11];
  const float* Wr1   = (const float*)d_in[12];
  const float* br1   = (const float*)d_in[13];
  float* out = (float*)d_out;

  // workspace layout (deterministic partial buffers, no fp atomics):
  // [wpart: 32*NN f32][w: NN f32][XsT: FF*NN bf16][Zpart: KSPLIT*NN*FF f32]
  float*  wpart = (float*)d_ws;
  float*  w     = wpart + (size_t)WSLICES * NN;
  __bf16* XsT   = (__bf16*)(w + NN);
  float*  Zpart = (float*)((char*)d_ws +
                  ((size_t)WSLICES * NN + NN) * sizeof(float) +
                  (size_t)FF * NN * sizeof(__bf16));

  wsum_part_kernel<<<dim3(NN / 256, WSLICES), 256, 0, stream>>>(T, theta, wpart);
  wreduce_kernel<<<NN / 256, 256, 0, stream>>>(wpart, w);
  scale_transpose_kernel<<<(NN * FF) / 256, 256, 0, stream>>>(X, w, XsT);
  gemm_wmma_kernel<<<dim3(NN / 128, KSPLIT), 256, 0, stream>>>(A, XsT, Zpart);
  tail_kernel<<<NN / 64, 256, 0, stream>>>(Zpart, Wd, bd, Wnf, bnf, Wm, bm,
                                           Wr0, br0, Wr1, br1, out);
}
